// RNNSequenceClassifier_489626272088
// MI455X (gfx1250) — matrix operand order repaired and checked by reading, not compiled, not run on hardware
//
#include <hip/hip_runtime.h>
#include <hip/hip_bf16.h>

// ---------------- problem constants (from reference) ----------------
#define B_SZ    64
#define S_LEN   128
#define T_TRI   10
#define E_DIM   300
#define D_DIM   100
#define H_DIM   256
#define C_CLS   3
#define IN_DIM  (E_DIM + 2 * D_DIM)   // 500
#define K_PAD   512                   // IN_DIM padded to WMMA K multiple
#define G4H     (4 * H_DIM)           // 1024
#define NTOK    (B_SZ * S_LEN)        // 8192
#define NPAIR   (NTOK * T_TRI)        // 81920 (token,triple) pairs
#define K2D     (2 * D_DIM)           // 200
#define K2D_PAD 224                   // 200 padded to K multiple of 32
#define MD_PAD  112                   // 100 padded to M multiple of 16

// ---------------- WMMA fragment types / helpers ----------------
typedef __attribute__((ext_vector_type(16))) _Float16 v16h;
typedef __attribute__((ext_vector_type(8)))  _Float16 v8h;
typedef __attribute__((ext_vector_type(8)))  float    v8f;

union HalfFrag { v16h v; v8h h2[2]; };

// A matrix 16x32 f16, row-major source [16, lda] (lda in halves, 16B aligned rows)
// Per ISA 7.12.2: lane m=lane&15, kh=lane>>4; VGPR0..3 hold K = kh*8 .. kh*8+7,
// VGPR4..7 hold K = 16+kh*8 .. 16+kh*8+7  (two contiguous 8-half runs).
__device__ inline v16h load_a_frag(const _Float16* __restrict__ A, int lda) {
  const int lane = threadIdx.x & 31;
  const int m  = lane & 15;
  const int kh = lane >> 4;
  HalfFrag f;
  f.h2[0] = *(const v8h*)(A + (size_t)m * lda + kh * 8);
  f.h2[1] = *(const v8h*)(A + (size_t)m * lda + 16 + kh * 8);
  return f.v;
}

// B matrix 32x16 f16 from W stored row-major [N, ldb] over K (B[k][n] = W[n][k]).
// lane n=lane&15, kh=lane>>4; elems 0..7 hold K=kh*8..kh*8+7, elems 8..15 hold K=16+kh*8..16+kh*8+7.
__device__ inline v16h load_b_frag(const _Float16* __restrict__ W, int ldb) {
  const int lane = threadIdx.x & 31;
  const int n  = lane & 15;
  const int kh = lane >> 4;
  HalfFrag f;
  f.h2[0] = *(const v8h*)(W + (size_t)n * ldb + kh * 8);
  f.h2[1] = *(const v8h*)(W + (size_t)n * ldb + 16 + kh * 8);
  return f.v;
}

__device__ inline v8f wmma16x16x32(v16h a, v16h b, v8f c) {
  return __builtin_amdgcn_wmma_f32_16x16x32_f16(false, a, false, b, (short)0, c,
                                                false, false);
}

__device__ inline float sigmoidf_(float x) { return 1.0f / (1.0f + __expf(-x)); }

// ---------------- K_conv: f32 weights -> f16 (padded) ----------------
__global__ __launch_bounds__(256)
void k_conv_weights(const float* __restrict__ Wih, const float* __restrict__ Whh,
                    const float* __restrict__ Went,
                    _Float16* __restrict__ WihH, _Float16* __restrict__ WhhH,
                    _Float16* __restrict__ WentH) {
  int i = blockIdx.x * blockDim.x + threadIdx.x;
  if (i < G4H * K_PAD) {
    int n = i / K_PAD, k = i - n * K_PAD;
    WihH[i] = (_Float16)((k < IN_DIM) ? Wih[(size_t)n * IN_DIM + k] : 0.0f);
  }
  if (i < G4H * H_DIM) {
    WhhH[i] = (_Float16)Whh[i];
  }
  if (i < MD_PAD * K2D_PAD) {
    int m = i / K2D_PAD, k = i - m * K2D_PAD;
    WentH[i] = (_Float16)((m < D_DIM && k < K2D) ? Went[(size_t)m * K2D + k] : 0.0f);
  }
}

// ---------------- k0a: stage HT = [eh | et] as f16 [NPAIR, K2D_PAD] ----------------
// one block of 224 threads per (token,triple) pair; thread = k index
__global__ __launch_bounds__(224)
void k0a_stage_ht(const int* __restrict__ triples,    // [B,S,T,3]
                  const float* __restrict__ embEnt,   // [VE,D]
                  _Float16* __restrict__ HT)          // [NPAIR, K2D_PAD]
{
  const int p = blockIdx.x;
  const int k = threadIdx.x;
  const int* tri = triples + (size_t)p * 3;
  float v = 0.0f;
  if (k < D_DIM)       v = embEnt[(size_t)tri[0] * D_DIM + k];
  else if (k < K2D)    v = embEnt[(size_t)tri[1] * D_DIM + (k - D_DIM)];
  HT[(size_t)p * K2D_PAD + k] = (_Float16)v;
}

// ---------------- k0b: WMMA GEMM tr = Went @ HT^T, fused tanh*er reduce -> e[p] ----
// block = 256 threads (8 waves); wave handles full M (7 tiles, d=0..111) x 16 pairs.
// W_ent f16 staged once in LDS; B-fragment (pairs) reused across the 7 M-tiles.
__global__ __launch_bounds__(256)
void k0b_graph_scores(const _Float16* __restrict__ HT,     // [NPAIR, K2D_PAD]
                      const _Float16* __restrict__ WentH,  // [MD_PAD, K2D_PAD]
                      const int* __restrict__ triples,     // [B,S,T,3]
                      const float* __restrict__ embRel,    // [VR,D]
                      float* __restrict__ eOut)            // [NPAIR]
{
  __shared__ __align__(16) _Float16 sh_went[MD_PAD * K2D_PAD];   // 49 KB

  const int tid  = threadIdx.x;
  const int wave = tid >> 5;
  const int lane = tid & 31;
  const int am   = lane & 15;
  const int akh  = lane >> 4;
  const int pBase = blockIdx.x * 128 + wave * 16;

  for (int i = tid; i < MD_PAD * K2D_PAD; i += 256) sh_went[i] = WentH[i];
  __syncthreads();

  v8f acc[7] = {};
  #pragma unroll
  for (int k = 0; k < K2D_PAD; k += 32) {
    v16h bf = load_b_frag(HT + (size_t)pBase * K2D_PAD + k, K2D_PAD);
    #pragma unroll
    for (int t = 0; t < 7; ++t) {
      HalfFrag af;                                                  // ds_load_b128
      af.h2[0] = *(const v8h*)(&sh_went[(t * 16 + am) * K2D_PAD + k + akh * 8]);
      af.h2[1] = *(const v8h*)(&sh_went[(t * 16 + am) * K2D_PAD + k + 16 + akh * 8]);
      acc[t] = wmma16x16x32(af.v, bf, acc[t]);
    }
  }

  // e[p] = sum_d tanh(tr[d,p]) * er[p][d]; C-layout: n=lane&15, m = t*16 + kh*8 + r
  const int p = pBase + (lane & 15);
  const int tri2 = triples[(size_t)p * 3 + 2];
  const float* erow = embRel + (size_t)tri2 * D_DIM;
  float part = 0.0f;
  #pragma unroll
  for (int t = 0; t < 7; ++t) {
    #pragma unroll
    for (int r = 0; r < 8; ++r) {
      const int d = t * 16 + akh * 8 + r;
      if (d < D_DIM) part += tanhf(acc[t][r]) * erow[d];
    }
  }
  part += __shfl_xor(part, 16);      // combine the two M-halves (same column p)
  if (akh == 0) eOut[p] = part;
}

// ---------------- k0c: softmax over triples + graph vec + emit t_in f16 ----------
// one block of 128 threads per token
__global__ __launch_bounds__(128)
void k0c_token_emit(const int* __restrict__ inputs,    // [B,S]
                    const int* __restrict__ id2,       // [B,S,T]
                    const float* __restrict__ emb,     // [VW,E]
                    const float* __restrict__ eIn,     // [NPAIR]
                    const _Float16* __restrict__ HT,   // [NPAIR, K2D_PAD]
                    _Float16* __restrict__ tin)        // [NTOK, K_PAD]
{
  __shared__ float sh_alpha[T_TRI];
  __shared__ int   sh_valid;

  const int tok = blockIdx.x;
  const int tid = threadIdx.x;

  if (tid == 0) {
    float ev[T_TRI];
    float mx = -3.0e38f;
    #pragma unroll
    for (int t = 0; t < T_TRI; ++t) {
      ev[t] = eIn[(size_t)tok * T_TRI + t];
      mx = fmaxf(mx, ev[t]);
    }
    float s = 0.0f;
    #pragma unroll
    for (int t = 0; t < T_TRI; ++t) { ev[t] = __expf(ev[t] - mx); s += ev[t]; }
    const float inv = 1.0f / s;
    #pragma unroll
    for (int t = 0; t < T_TRI; ++t) sh_alpha[t] = ev[t] * inv;
    int v = 0;
    #pragma unroll
    for (int t = 0; t < T_TRI; ++t) v |= (id2[(size_t)tok * T_TRI + t] != -1) ? 1 : 0;
    sh_valid = v;
  }
  __syncthreads();

  _Float16* o = tin + (size_t)tok * K_PAD;
  const int widx = inputs[tok];
  for (int i = tid; i < E_DIM; i += 128)
    o[i] = (_Float16)emb[(size_t)widx * E_DIM + i];
  const int valid = sh_valid;
  for (int i = tid; i < K2D; i += 128) {
    float g = 0.0f;
    if (valid) {
      #pragma unroll
      for (int t = 0; t < T_TRI; ++t)
        g += sh_alpha[t] * (float)HT[((size_t)tok * T_TRI + t) * K2D_PAD + i];
    }
    o[E_DIM + i] = (_Float16)g;
  }
  for (int i = IN_DIM + tid; i < K_PAD; i += 128) o[i] = (_Float16)0.0f;
}

// ---------------- K1: input projection GEMM, WMMA f16 ----------------
// xg[s][b][n] = t_in[tok,:] . W_ih[n,:] + b[n]   (M=8192, N=1024, K=512)
__global__ __launch_bounds__(256)
void k1_input_proj(const _Float16* __restrict__ Ain,   // [NTOK, K_PAD], tok = b*S+s
                   const _Float16* __restrict__ Wih,   // [G4H, K_PAD]
                   const float* __restrict__ bias,     // [G4H]
                   _Float16* __restrict__ xg)          // [S, B, G4H] (f16)
{
  const int wave  = threadIdx.x >> 5;
  const int mBase = blockIdx.y * 128 + wave * 16;
  const int nBase = blockIdx.x * 64;
  const _Float16* aRow = Ain + (size_t)mBase * K_PAD;

  v8f acc[4] = {};
  #pragma unroll 2
  for (int k = 0; k < K_PAD; k += 32) {
    __builtin_prefetch(aRow + k + 64, 0, 1);           // global_prefetch_b8
    v16h a = load_a_frag(aRow + k, K_PAD);
    #pragma unroll
    for (int t = 0; t < 4; ++t) {
      v16h b = load_b_frag(Wih + (size_t)(nBase + t * 16) * K_PAD + k, K_PAD);
      acc[t] = wmma16x16x32(a, b, acc[t]);
    }
  }

  const int lane = threadIdx.x & 31;
  const int n0   = lane & 15;
  const int mb0  = (lane >> 4) * 8;
  #pragma unroll
  for (int t = 0; t < 4; ++t) {
    const int n = nBase + t * 16 + n0;
    const float bn = bias[n];
    #pragma unroll
    for (int r = 0; r < 8; ++r) {
      const int tok = mBase + mb0 + r;                 // tok = b*S + s
      const int b = tok >> 7, s = tok & (S_LEN - 1);
      xg[((size_t)s * B_SZ + b) * G4H + n] = (_Float16)(acc[t][r] + bn);
    }
  }
}

// ---------------- K2: persistent LSTM recurrence (WMMA per step) ----------------
// 4 blocks x 256 threads; each block owns 16 batch rows; loops s=0..127 with
// workgroup barriers. h (f16), c (f32), g (f16) live in LDS (56 KB).
__global__ __launch_bounds__(256)
void k2_lstm(const _Float16* __restrict__ xg,    // [S, B, G4H] f16
             const _Float16* __restrict__ WhhH,  // [G4H, H_DIM] f16
             float* __restrict__ out)            // [B, S, H] f32
{
  __shared__ __align__(16) _Float16 sh_h[16 * H_DIM];   // 8 KB
  __shared__ __align__(16) _Float16 sh_g[16 * G4H];     // 32 KB
  __shared__ __align__(16) float    sh_c[16 * H_DIM];   // 16 KB

  const int b0   = blockIdx.x * 16;
  const int tid  = threadIdx.x;
  const int wave = tid >> 5;
  const int lane = tid & 31;
  const int am   = lane & 15;
  const int akh  = lane >> 4;
  const int n0   = lane & 15;
  const int mb0  = (lane >> 4) * 8;
  const int nW   = wave * 128;

  for (int i = tid; i < 16 * H_DIM; i += 256) { sh_h[i] = (_Float16)0.0f; sh_c[i] = 0.0f; }
  __syncthreads();

  for (int s = 0; s < S_LEN; ++s) {
    const _Float16* xgRow = xg + (size_t)s * B_SZ * G4H;
    v8f acc[8] = {};
    #pragma unroll
    for (int k = 0; k < H_DIM; k += 32) {
      HalfFrag af;                                                  // ds_load_b128
      af.h2[0] = *(const v8h*)(&sh_h[am * H_DIM + k + akh * 8]);
      af.h2[1] = *(const v8h*)(&sh_h[am * H_DIM + k + 16 + akh * 8]);
      #pragma unroll
      for (int t = 0; t < 8; ++t) {
        v16h bf = load_b_frag(WhhH + (size_t)(nW + t * 16) * H_DIM + k, H_DIM);
        acc[t] = wmma16x16x32(af.v, bf, acc[t]);
      }
    }
    #pragma unroll
    for (int t = 0; t < 8; ++t) {
      const int n = nW + t * 16 + n0;
      #pragma unroll
      for (int r = 0; r < 8; ++r) {
        const int mb = mb0 + r;
        float g = acc[t][r] + (float)xgRow[(size_t)(b0 + mb) * G4H + n];
        sh_g[mb * G4H + n] = (_Float16)g;
      }
    }
    __syncthreads();

    for (int i = tid; i < 16 * H_DIM; i += 256) {
      const int mb = i >> 8;
      const int hh = i & (H_DIM - 1);
      const float gi = (float)sh_g[mb * G4H + hh];
      const float gf = (float)sh_g[mb * G4H + H_DIM + hh];
      const float gg = (float)sh_g[mb * G4H + 2 * H_DIM + hh];
      const float go = (float)sh_g[mb * G4H + 3 * H_DIM + hh];
      float c = sigmoidf_(gf) * sh_c[i] + sigmoidf_(gi) * tanhf(gg);
      float h = sigmoidf_(go) * tanhf(c);
      sh_c[i] = c;
      sh_h[mb * H_DIM + hh] = (_Float16)h;
      out[((size_t)(b0 + mb) * S_LEN + s) * H_DIM + hh] = h;
    }
    __syncthreads();
  }
}

// ---------------- K3: masked attention pooling + logits ----------------
__global__ __launch_bounds__(128)
void k3_attn_out(const float* __restrict__ out,    // [B,S,H]
                 const int* __restrict__ lengths,  // [B]
                 const float* __restrict__ attn_w, const float* __restrict__ attn_b,
                 const float* __restrict__ out_w,  const float* __restrict__ out_b,
                 float* __restrict__ logits)       // [B,C]
{
  __shared__ float sh_sc[S_LEN];
  __shared__ float sh_att[H_DIM];
  __shared__ float sh_red[128];

  const int b = blockIdx.x, tid = threadIdx.x;
  const int len = lengths[b];

  {
    const float* o = out + ((size_t)b * S_LEN + tid) * H_DIM;
    float acc = attn_b[0];
    #pragma unroll 4
    for (int h = 0; h < H_DIM; ++h) acc += o[h] * attn_w[h];
    sh_sc[tid] = (tid < len) ? acc : -1.0e9f;
  }
  __syncthreads();

  sh_red[tid] = sh_sc[tid];
  __syncthreads();
  for (int off = 64; off > 0; off >>= 1) {
    if (tid < off) sh_red[tid] = fmaxf(sh_red[tid], sh_red[tid + off]);
    __syncthreads();
  }
  const float mx = sh_red[0];
  __syncthreads();
  const float ex = __expf(sh_sc[tid] - mx);
  sh_red[tid] = ex;
  __syncthreads();
  for (int off = 64; off > 0; off >>= 1) {
    if (tid < off) sh_red[tid] += sh_red[tid + off];
    __syncthreads();
  }
  const float inv = 1.0f / sh_red[0];
  __syncthreads();
  sh_sc[tid] = ex * inv;
  __syncthreads();

  for (int h = tid; h < H_DIM; h += 128) {
    float a = 0.0f;
    for (int s = 0; s < S_LEN; ++s)
      a += sh_sc[s] * out[((size_t)b * S_LEN + s) * H_DIM + h];
    sh_att[h] = a;
  }
  __syncthreads();

  if (tid < C_CLS) {
    float acc = out_b[tid];
    #pragma unroll 4
    for (int h = 0; h < H_DIM; ++h) acc += sh_att[h] * out_w[(size_t)tid * H_DIM + h];
    logits[(size_t)b * C_CLS + tid] = acc;
  }
}

// ---------------- launch ----------------
extern "C" void kernel_launch(void* const* d_in, const int* in_sizes, int n_in,
                              void* d_out, int out_size, void* d_ws, size_t ws_size,
                              hipStream_t stream) {
  (void)in_sizes; (void)n_in; (void)out_size; (void)ws_size;

  const int*   inputs  = (const int*)d_in[0];
  /* d_in[1] = context: unused by the reference */
  const int*   triples = (const int*)d_in[2];
  const int*   lengths = (const int*)d_in[3];
  const int*   id2     = (const int*)d_in[4];
  const float* emb     = (const float*)d_in[5];
  const float* embEnt  = (const float*)d_in[6];
  const float* embRel  = (const float*)d_in[7];
  const float* Went    = (const float*)d_in[8];
  const float* Wih     = (const float*)d_in[9];
  const float* Whh     = (const float*)d_in[10];
  const float* bLstm   = (const float*)d_in[11];
  const float* attnW   = (const float*)d_in[12];
  const float* attnB   = (const float*)d_in[13];
  const float* outW    = (const float*)d_in[14];
  const float* outB    = (const float*)d_in[15];
  float* logits = (float*)d_out;

  // workspace layout (all offsets 512B-aligned by construction); total ~71 MB
  char* ws = (char*)d_ws;
  _Float16* tin   = (_Float16*)ws; ws += (size_t)NTOK  * K_PAD   * sizeof(_Float16); // 8 MB
  _Float16* WihH  = (_Float16*)ws; ws += (size_t)G4H   * K_PAD   * sizeof(_Float16); // 1 MB
  _Float16* WhhH  = (_Float16*)ws; ws += (size_t)G4H   * H_DIM   * sizeof(_Float16); // 0.5 MB
  _Float16* xg    = (_Float16*)ws; ws += (size_t)NTOK  * G4H     * sizeof(_Float16); // 16 MB
  float*    hseq  = (float*)ws;    ws += (size_t)B_SZ * S_LEN * H_DIM * sizeof(float); // 8 MB
  _Float16* HT    = (_Float16*)ws; ws += (size_t)NPAIR * K2D_PAD * sizeof(_Float16); // 36.7 MB
  _Float16* WentH = (_Float16*)ws; ws += (size_t)MD_PAD * K2D_PAD * sizeof(_Float16); // 50 KB
  float*    eGlob = (float*)ws;    ws += (size_t)NPAIR * sizeof(float);              // 0.33 MB

  k_conv_weights<<<(G4H * K_PAD + 255) / 256, 256, 0, stream>>>(
      Wih, Whh, Went, WihH, WhhH, WentH);
  k0a_stage_ht<<<NPAIR, 224, 0, stream>>>(triples, embEnt, HT);
  k0b_graph_scores<<<NPAIR / 128, 256, 0, stream>>>(HT, WentH, triples, embRel, eGlob);
  k0c_token_emit<<<NTOK, 128, 0, stream>>>(inputs, id2, emb, eGlob, HT, tin);
  k1_input_proj<<<dim3(G4H / 64, NTOK / 128), 256, 0, stream>>>(tin, WihH, bLstm, xg);
  k2_lstm<<<B_SZ / 16, 256, 0, stream>>>(xg, WhhH, hseq);
  k3_attn_out<<<B_SZ, 128, 0, stream>>>(hseq, lengths, attnW, attnB, outW, outB, logits);
}
